// PointNetFeaturePropagation_14078902796587
// MI455X (gfx1250) — matrix operand order repaired and checked by reading, not compiled, not run on hardware
//
#include <hip/hip_runtime.h>
#include <hip/hip_bf16.h>
#include <cstdint>

// ---------------------------------------------------------------------------
// PointNet Feature Propagation, MI455X (gfx1250, wave32, WMMA bf16 16x16x32)
//   B=16, N=4096, M=1024, D1=128, D2=256, C0=384, H0=256, H1=128
// GEMMs (~17.2 GFLOP) on v_wmma_f32_16x16x32_bf16, BN folded into W/bias.
// Activation tiles staged Global->LDS via the async path (ASYNCcnt),
// double-buffered over K so copies overlap WMMA. All intermediates L2-resident.
// ---------------------------------------------------------------------------

#define BB 16
#define NPTS 4096
#define MPTS 1024
#define D1C 128
#define D2C 256
#define C0 384
#define H0 256
#define H1 128
#define KC 128      // K-chunk staged per step
#define KCP 136     // padded LDS row stride (elems): 272B -> conflict-free frags

typedef __attribute__((ext_vector_type(16))) __bf16 v16bf;
typedef __attribute__((ext_vector_type(8)))  float  v8f;
typedef int v4i_vec __attribute__((vector_size(4 * sizeof(int))));

#define AS_GLOBAL __attribute__((address_space(1)))
#define AS_SHARED __attribute__((address_space(3)))

union Frag32B { v16bf v; uint4 u4[2]; };

#if defined(__has_builtin)
# if __has_builtin(__builtin_amdgcn_global_load_async_to_lds_b128)
#  define HAVE_ASYNC_LDS 1
# endif
#endif
#ifndef HAVE_ASYNC_LDS
# define HAVE_ASYNC_LDS 0
#endif

__device__ __forceinline__ uint16_t f2bf(float f) {
  uint32_t u = __builtin_bit_cast(uint32_t, f);
  u += 0x7FFFu + ((u >> 16) & 1u);          // round-to-nearest-even
  return (uint16_t)(u >> 16);
}

__device__ __forceinline__ void wait_async_lds() {
#if HAVE_ASYNC_LDS
# if __has_builtin(__builtin_amdgcn_s_wait_asynccnt)
  __builtin_amdgcn_s_wait_asynccnt(0);
# else
  asm volatile("s_wait_asynccnt 0x0" ::: "memory");
# endif
#endif
}

// Stage one K-chunk (64 rows x 128 elems bf16) of the activation tile into LDS.
// 1024 x 16B transfers, 4 per thread (blockDim == 256).
template <int K>
__device__ __forceinline__ void stage_chunk(const uint16_t* __restrict__ src, int c,
                                            uint16_t* __restrict__ dst) {
  const int tid = threadIdx.x;
#pragma unroll
  for (int i = 0; i < 4; ++i) {
    const int idx = tid + i * 256;
    const int r = idx >> 4;          // point row 0..63
    const int s = idx & 15;          // 16B segment within the 256B chunk row
    const uint16_t* g = src + (size_t)r * K + c * KC + s * 8;
    uint16_t* l = dst + r * KCP + s * 8;
#if HAVE_ASYNC_LDS
    __builtin_amdgcn_global_load_async_to_lds_b128(
        (AS_GLOBAL v4i_vec*)g, (AS_SHARED v4i_vec*)l, 0, 0);
#else
    *reinterpret_cast<uint4*>(l) = *reinterpret_cast<const uint4*>(g);
#endif
  }
}

// ---------------------------------------------------------------------------
// Fold BatchNorm into conv weights/bias; convert weights to bf16.
// ---------------------------------------------------------------------------
__global__ void prep_weights(const float* __restrict__ w0, const float* __restrict__ b0,
                             const float* __restrict__ g0, const float* __restrict__ be0,
                             const float* __restrict__ m0, const float* __restrict__ v0,
                             const float* __restrict__ w1, const float* __restrict__ b1,
                             const float* __restrict__ g1, const float* __restrict__ be1,
                             const float* __restrict__ m1, const float* __restrict__ v1,
                             uint16_t* __restrict__ W0p, float* __restrict__ bias0,
                             uint16_t* __restrict__ W1p, float* __restrict__ bias1) {
  const int tid = blockIdx.x * blockDim.x + threadIdx.x;
  const int stride = gridDim.x * blockDim.x;
  for (int i = tid; i < H0 * C0; i += stride) {
    int o = i / C0;
    float s = g0[o] * rsqrtf(v0[o] + 1e-5f);
    W0p[i] = f2bf(w0[i] * s);
  }
  for (int i = tid; i < H1 * H0; i += stride) {
    int o = i / H0;
    float s = g1[o] * rsqrtf(v1[o] + 1e-5f);
    W1p[i] = f2bf(w1[i] * s);
  }
  if (tid < H0) {
    float s = g0[tid] * rsqrtf(v0[tid] + 1e-5f);
    bias0[tid] = s * (b0[tid] - m0[tid]) + be0[tid];
  }
  if (tid < H1) {
    float s = g1[tid] * rsqrtf(v1[tid] + 1e-5f);
    bias1[tid] = s * (b1[tid] - m1[tid]) + be1[tid];
  }
}

// ---------------------------------------------------------------------------
// Brute-force 3-NN + inverse-distance weights; xyz2 of the batch in LDS (12KB).
// ---------------------------------------------------------------------------
__global__ void three_nn_kernel(const float* __restrict__ xyz1, const float* __restrict__ xyz2,
                                int* __restrict__ idx_out, float* __restrict__ wgt_out) {
  __shared__ float s2[MPTS * 3];
  const int b = blockIdx.y;
  const int n = blockIdx.x * blockDim.x + threadIdx.x;
  const float* x2 = xyz2 + (size_t)b * MPTS * 3;
  for (int i = threadIdx.x; i < MPTS * 3; i += blockDim.x) s2[i] = x2[i];
  __syncthreads();
  const float* p = xyz1 + ((size_t)b * NPTS + n) * 3;
  const float px = p[0], py = p[1], pz = p[2];
  float d0 = 1e30f, d1 = 1e30f, d2 = 1e30f;
  int   i0 = 0, i1 = 0, i2 = 0;
  for (int m = 0; m < MPTS; ++m) {
    float dx = px - s2[3 * m + 0];
    float dy = py - s2[3 * m + 1];
    float dz = pz - s2[3 * m + 2];
    float d = dx * dx + dy * dy + dz * dz;
    if (d < d2) {
      if (d < d1) {
        d2 = d1; i2 = i1;
        if (d < d0) { d1 = d0; i1 = i0; d0 = d; i0 = m; }
        else        { d1 = d;  i1 = m; }
      } else { d2 = d; i2 = m; }
    }
  }
  float r0 = 1.f / (d0 + 1e-8f), r1 = 1.f / (d1 + 1e-8f), r2 = 1.f / (d2 + 1e-8f);
  float rs = 1.f / (r0 + r1 + r2);
  size_t o = ((size_t)b * NPTS + n) * 3;
  idx_out[o] = i0; idx_out[o + 1] = i1; idx_out[o + 2] = i2;
  wgt_out[o] = r0 * rs; wgt_out[o + 1] = r1 * rs; wgt_out[o + 2] = r2 * rs;
}

// ---------------------------------------------------------------------------
// Interpolate feat2 by 3-NN weights, concat feat1, write x transposed
// [B][N][C0] bf16 so WMMA B-fragments are two contiguous 16B runs.
// ---------------------------------------------------------------------------
__global__ void interp_kernel(const float* __restrict__ feat1, const float* __restrict__ feat2,
                              const int* __restrict__ idx, const float* __restrict__ wgt,
                              uint16_t* __restrict__ xT) {
  const int b = blockIdx.y;
  const int lane = threadIdx.x & 31;
  const int wave = threadIdx.x >> 5;
  const int n = blockIdx.x * 32 + lane;
  const size_t o3 = ((size_t)b * NPTS + n) * 3;
  const int i0 = idx[o3], i1 = idx[o3 + 1], i2 = idx[o3 + 2];
  const float w0 = wgt[o3], w1 = wgt[o3 + 1], w2 = wgt[o3 + 2];
  const float* f2 = feat2 + (size_t)b * D2C * MPTS;
  const float* f1 = feat1 + (size_t)b * D1C * NPTS;
  uint16_t* xrow = xT + ((size_t)b * NPTS + n) * C0;
  const int cbeg = wave * 48;
  for (int c = cbeg; c < cbeg + 48; ++c) {
    float v;
    if (c < D2C) {
      const float* row = f2 + (size_t)c * MPTS;
      v = w0 * row[i0] + w1 * row[i1] + w2 * row[i2];
    } else {
      v = f1[(size_t)(c - D2C) * NPTS + n];
    }
    xrow[c] = f2bf(v);
  }
}

// ---------------------------------------------------------------------------
// Fused GEMM + bias + ReLU:  Y = relu(W' X + b')
//   X   : [B][N][K] bf16 (activations, point-major)
//   W   : [COUT][K] bf16 (BN-folded)
// Block = 8 waves; wave owns 16 outch x 64 pts (4 accumulators, A reused 4x).
// Activation tile (64 x K) staged to LDS in 128-elem K-chunks, double-buffered
// with async Global->LDS copies; all 8 waves share it (8x L2 traffic cut).
// A frag: row = o0+(lane&15), K chunks [kb..kb+7],[kb+16..kb+23], kb=(lane>>4)*8.
// B frag: col = p+(lane&15),  K chunks [kb..kb+7],[kb+16..kb+23] as A (two ds_b128).
// C/D:    vgpr r, lane l -> M = r + (l>>4)*8, N = l&15.
// ---------------------------------------------------------------------------
template <int K, int NKC, int COUT, bool F32OUT>
__global__ void gemm_wmma(const uint16_t* __restrict__ W, const float* __restrict__ bias,
                          const uint16_t* __restrict__ X,
                          uint16_t* __restrict__ Ybf, float* __restrict__ Yf) {
  __shared__ __align__(16) uint16_t sx[2][64 * KCP];
  const int b = blockIdx.z;
  const int wave = threadIdx.x >> 5, lane = threadIdx.x & 31;
  const int o0 = blockIdx.y * 128 + wave * 16;
  const int p0 = blockIdx.x * 64;
  const int lo = lane & 15, hi = lane >> 4;

  const uint16_t* xblk = X + ((size_t)b * NPTS + p0) * K;
  const uint16_t* wrow = W + (size_t)(o0 + lo) * K;

  // C init = folded-BN bias broadcast along N
  v8f acc[4];
  {
    const float4* bp = reinterpret_cast<const float4*>(bias + o0 + hi * 8);
    float4 b01 = bp[0], b23 = bp[1];
    v8f c;
    c[0] = b01.x; c[1] = b01.y; c[2] = b01.z; c[3] = b01.w;
    c[4] = b23.x; c[5] = b23.y; c[6] = b23.z; c[7] = b23.w;
    acc[0] = c; acc[1] = c; acc[2] = c; acc[3] = c;
  }

  stage_chunk<K>(xblk, 0, &sx[0][0]);
  wait_async_lds();
  __syncthreads();

  for (int c = 0; c < NKC; ++c) {
    if (c + 1 < NKC) stage_chunk<K>(xblk, c + 1, &sx[(c + 1) & 1][0]);
    const uint16_t* sbuf = &sx[c & 1][0];
#pragma unroll
    for (int kk = 0; kk < KC; kk += 32) {
      Frag32B a;
      const uint4* ap = reinterpret_cast<const uint4*>(wrow + c * KC + kk + hi * 8);
      a.u4[0] = ap[0];   // K = +kb .. +kb+7
      a.u4[1] = ap[2];   // K = +kb+16 .. +kb+23
#pragma unroll
      for (int t = 0; t < 4; ++t) {
        Frag32B bf;
        const uint4* bp =
            reinterpret_cast<const uint4*>(sbuf + (t * 16 + lo) * KCP + kk + hi * 8);
        bf.u4[0] = bp[0];
        bf.u4[1] = bp[2];
        acc[t] = __builtin_amdgcn_wmma_f32_16x16x32_bf16(false, a.v, false, bf.v,
                                                         (short)0, acc[t], false, false);
      }
    }
    if (c + 1 < NKC) {
      wait_async_lds();
      __syncthreads();
    }
  }

#pragma unroll
  for (int t = 0; t < 4; ++t) {
    const int n = p0 + t * 16 + lo;
    if constexpr (F32OUT) {
#pragma unroll
      for (int r = 0; r < 8; ++r) {
        const int o = o0 + hi * 8 + r;
        Yf[((size_t)b * COUT + o) * NPTS + n] = fmaxf(acc[t][r], 0.f);
      }
    } else {
      uint16_t* orow = Ybf + ((size_t)b * NPTS + n) * COUT + o0 + hi * 8;
      uint32_t pk[4];
#pragma unroll
      for (int r = 0; r < 4; ++r) {
        float f0 = fmaxf(acc[t][2 * r], 0.f);
        float f1 = fmaxf(acc[t][2 * r + 1], 0.f);
        pk[r] = (uint32_t)f2bf(f0) | ((uint32_t)f2bf(f1) << 16);
      }
      *reinterpret_cast<uint4*>(orow) = make_uint4(pk[0], pk[1], pk[2], pk[3]);
    }
  }
}

// ---------------------------------------------------------------------------
extern "C" void kernel_launch(void* const* d_in, const int* in_sizes, int n_in,
                              void* d_out, int out_size, void* d_ws, size_t ws_size,
                              hipStream_t stream) {
  const float* xyz1  = (const float*)d_in[0];
  const float* xyz2  = (const float*)d_in[1];
  const float* feat1 = (const float*)d_in[2];
  const float* feat2 = (const float*)d_in[3];
  const float* w0  = (const float*)d_in[4];
  const float* b0  = (const float*)d_in[5];
  const float* g0  = (const float*)d_in[6];
  const float* be0 = (const float*)d_in[7];
  const float* m0  = (const float*)d_in[8];
  const float* v0  = (const float*)d_in[9];
  const float* w1  = (const float*)d_in[10];
  const float* b1  = (const float*)d_in[11];
  const float* g1  = (const float*)d_in[12];
  const float* be1 = (const float*)d_in[13];
  const float* m1  = (const float*)d_in[14];
  const float* v1  = (const float*)d_in[15];

  char* ws = (char*)d_ws;
  size_t off = 0;
  auto carve = [&](size_t bytes) -> void* {
    off = (off + 255) & ~(size_t)255;
    void* p = ws + off;
    off += bytes;
    return p;
  };
  uint16_t* W0p   = (uint16_t*)carve((size_t)H0 * C0 * 2);
  float*    bias0 = (float*)   carve((size_t)H0 * 4);
  uint16_t* W1p   = (uint16_t*)carve((size_t)H1 * H0 * 2);
  float*    bias1 = (float*)   carve((size_t)H1 * 4);
  int*      nnidx = (int*)     carve((size_t)BB * NPTS * 3 * 4);
  float*    nnwgt = (float*)   carve((size_t)BB * NPTS * 3 * 4);
  uint16_t* xT    = (uint16_t*)carve((size_t)BB * NPTS * C0 * 2);
  uint16_t* y0t   = (uint16_t*)carve((size_t)BB * NPTS * H0 * 2);
  (void)ws_size; (void)in_sizes; (void)n_in; (void)out_size;

  prep_weights<<<dim3(64), dim3(256), 0, stream>>>(
      w0, b0, g0, be0, m0, v0, w1, b1, g1, be1, m1, v1, W0p, bias0, W1p, bias1);

  three_nn_kernel<<<dim3(NPTS / 256, BB), dim3(256), 0, stream>>>(xyz1, xyz2, nnidx, nnwgt);

  interp_kernel<<<dim3(NPTS / 32, BB), dim3(256), 0, stream>>>(feat1, feat2, nnidx, nnwgt, xT);

  gemm_wmma<C0, C0 / KC, H0, false><<<dim3(NPTS / 64, H0 / 128, BB), dim3(256), 0, stream>>>(
      W0p, bias0, xT, y0t, nullptr);

  gemm_wmma<H0, H0 / KC, H1, true><<<dim3(NPTS / 64, 1, BB), dim3(256), 0, stream>>>(
      W1p, bias1, y0t, nullptr, (float*)d_out);
}
